// TimeRecursive_Mamba_52956946760430
// MI455X (gfx1250) — matrix operand order repaired and checked by reading, not compiled, not run on hardware
//
#include <hip/hip_runtime.h>
#include <hip/hip_bf16.h>
#include <math.h>

// ---------------------------------------------------------------------------
// TimeRecursive Mamba + Conformer for MI455X (gfx1250, wave32, WMMA)
// ---------------------------------------------------------------------------

typedef __attribute__((ext_vector_type(16))) _Float16 v16h;
typedef __attribute__((ext_vector_type(8)))  float    v8f;

#define BSZ      2
#define LSEQ     512
#define CH       128      // C
#define DI_      256      // EXPAND*C
#define DTR      8        // DT_RANK
#define INNER_   256      // 2*C
#define KDW      31
#define XDB_LD   80       // padded row stride for xdb (max 8+2*32=72)

__device__ __forceinline__ float silu_f(float x) { return x / (1.f + __expf(-x)); }
__device__ __forceinline__ float sigm_f(float x) { return 1.f / (1.f + __expf(-x)); }

// ---------------------------------------------------------------------------
// Weight pre-pack: f32 row-major W[K,N] -> f16 fragment-ordered, N padded.
// Packed index: (((ntile*(K/32) + kb)*32 + lane)*16 + h)
//   lane 0-15  -> n = ntile*16+lane,    k = kb*32 + h       (B frag K 0-15)
//   lane 16-31 -> n = ntile*16+lane-16, k = kb*32 + 16 + h  (B frag K 16-31)
// ---------------------------------------------------------------------------
__global__ void pack_w_k(const float* __restrict__ W, _Float16* __restrict__ Wp,
                         int K, int N, int total) {
  const int i = blockIdx.x * blockDim.x + threadIdx.x;
  if (i >= total) return;
  const int h    = i & 15;
  const int lane = (i >> 4) & 31;
  const int blk  = i >> 9;                 // ntile*(K/32) + kb
  const int kb   = blk % (K >> 5);
  const int nt   = blk / (K >> 5);
  const int n = nt * 16 + (lane & 15);
  const int k = kb * 32 + ((lane & 16) ? 16 : 0) + h;
  Wp[i] = (n < N) ? (_Float16)W[(size_t)k * N + n] : (_Float16)0.f;
}

// ---------------------------------------------------------------------------
// GEMM: C[M,Npad] = A[M,K] @ Wp + bias. One wave per 16x(16*NT) output strip:
// A fragment loaded once per K-step and shared by NT WMMAs (NT B fragments,
// each a single contiguous 32B v16h load from the packed layout).
// Requires: M % 16 == 0, K in {128,256}, Npad % (16*NT) == 0,
// lda/ldc multiples of 16, A rows 16B aligned. K-loop fully unrolled.
// ---------------------------------------------------------------------------
template <int K, int NT>
__global__ __launch_bounds__(32)
void gemm_wmma_k(const float* __restrict__ A, const _Float16* __restrict__ Wp,
                 const float* __restrict__ bias, float* __restrict__ Cmat,
                 int lda, int ldc) {
  const int lane  = threadIdx.x;
  const int m0    = blockIdx.y * 16;
  const int tn0   = blockIdx.x * NT;
  const int aksel = (lane & 16) ? 16 : 0;

  const float* __restrict__ Arow = A + (size_t)(m0 + (lane & 15)) * lda;
  const _Float16* __restrict__ Wlane = Wp + (size_t)lane * 16;

  v8f acc[NT];
#pragma unroll
  for (int j = 0; j < NT; ++j) acc[j] = (v8f){};

#pragma unroll
  for (int kb = 0; kb < K / 32; ++kb) {
    const int k0 = kb * 32;
    const float4 a0 = *(const float4*)(Arow + k0 + aksel);
    const float4 a1 = *(const float4*)(Arow + k0 + aksel + 4);
    const float4 a2 = *(const float4*)(Arow + k0 + aksel + 8);
    const float4 a3 = *(const float4*)(Arow + k0 + aksel + 12);
    v16h af;
    af[0]  = (_Float16)a0.x; af[1]  = (_Float16)a0.y;
    af[2]  = (_Float16)a0.z; af[3]  = (_Float16)a0.w;
    af[4]  = (_Float16)a1.x; af[5]  = (_Float16)a1.y;
    af[6]  = (_Float16)a1.z; af[7]  = (_Float16)a1.w;
    af[8]  = (_Float16)a2.x; af[9]  = (_Float16)a2.y;
    af[10] = (_Float16)a2.z; af[11] = (_Float16)a2.w;
    af[12] = (_Float16)a3.x; af[13] = (_Float16)a3.y;
    af[14] = (_Float16)a3.z; af[15] = (_Float16)a3.w;

    v16h bf[NT];
#pragma unroll
    for (int j = 0; j < NT; ++j)
      bf[j] = *(const v16h*)(Wlane + ((size_t)(tn0 + j) * (K / 32) + kb) * 512);
#pragma unroll
    for (int j = 0; j < NT; ++j)
      acc[j] = __builtin_amdgcn_wmma_f32_16x16x32_f16(false, af, false, bf[j],
                                                      (short)0, acc[j], false, false);
  }

  const int rofs = (lane & 16) ? 8 : 0;
#pragma unroll
  for (int j = 0; j < NT; ++j) {
    const int col  = (tn0 + j) * 16 + (lane & 15);
    const float bv = bias ? bias[col] : 0.f;
#pragma unroll
    for (int r = 0; r < 8; ++r)
      Cmat[(size_t)(m0 + r + rofs) * ldc + col] = acc[j][r] + bv;
  }
}

// ---------------------------------------------------------------------------
// Fused LayerNorm over C=128: one wave per row, 4 elems/lane, shfl_xor reduce.
// mode bit0: x += x2 before LN ; bit1: += res after LN ; bit2: relu
// ---------------------------------------------------------------------------
__global__ __launch_bounds__(32)
void ln_fused_k(const float* __restrict__ x, const float* __restrict__ x2,
                const float* __restrict__ w, const float* __restrict__ b,
                const float* __restrict__ res, float* __restrict__ y, int mode) {
  const int row  = blockIdx.x;
  const int lane = threadIdx.x;
  const size_t base = (size_t)row * CH;
  float v[4];
  float s = 0.f;
#pragma unroll
  for (int i = 0; i < 4; ++i) {
    const int c = lane + 32 * i;
    float t = x[base + c];
    if (mode & 1) t += x2[base + c];
    v[i] = t; s += t;
  }
#pragma unroll
  for (int off = 16; off >= 1; off >>= 1) s += __shfl_xor(s, off, 32);
  const float mean = s * (1.f / CH);
  float var = 0.f;
#pragma unroll
  for (int i = 0; i < 4; ++i) { const float d = v[i] - mean; var += d * d; }
#pragma unroll
  for (int off = 16; off >= 1; off >>= 1) var += __shfl_xor(var, off, 32);
  const float inv = rsqrtf(var * (1.f / CH) + 1e-5f);
#pragma unroll
  for (int i = 0; i < 4; ++i) {
    const int c = lane + 32 * i;
    float o = (v[i] - mean) * inv * w[c] + b[c];
    if (mode & 2) o += res[base + c];
    if (mode & 4) o = fmaxf(o, 0.f);
    y[base + c] = o;
  }
}

// Mamba causal depthwise conv (+bias +silu). xz row stride 512, first DI cols.
__global__ void causal_conv_silu_k(const float* __restrict__ xz,
                                   const float* __restrict__ cw,
                                   const float* __restrict__ cb,
                                   float* __restrict__ xc,
                                   int Lx, int dc, int total) {
  const int i = blockIdx.x * blockDim.x + threadIdx.x;
  if (i >= total) return;
  const int ch = i & (DI_ - 1);
  const int t  = i >> 8;           // b*Lx + l
  const int l  = t % Lx;
  const int bb = t / Lx;
  float acc = cb[ch];
  for (int k = 0; k < dc; ++k) {
    const int ll = l - (dc - 1) + k;
    if (ll >= 0) acc += xz[((size_t)(bb * Lx + ll)) * (2 * DI_) + ch] * cw[ch * dc + k];
  }
  xc[(size_t)t * DI_ + ch] = silu_f(acc);
}

// dt = softplus(xdb[:, :8] @ W_dt + b_dt)   (K=8: scalar path, not worth WMMA)
__global__ void dt_proj_k(const float* __restrict__ xdb, const float* __restrict__ Wdt,
                          const float* __restrict__ bdt, float* __restrict__ dt, int total) {
  const int i = blockIdx.x * blockDim.x + threadIdx.x;
  if (i >= total) return;
  const int d = i & (DI_ - 1);
  const int r = i >> 8;
  float acc = bdt[d];
#pragma unroll
  for (int j = 0; j < DTR; ++j) acc += xdb[(size_t)r * XDB_LD + j] * Wdt[j * DI_ + d];
  dt[i] = (acc > 20.f) ? acc : log1pf(__expf(acc));
}

// SSM scan: one wave per (b, d); state index s in lanes; serial over L.
__global__ __launch_bounds__(32)
void ssm_scan_k(const float* __restrict__ dt, const float* __restrict__ xdb,
                const float* __restrict__ xc, const float* __restrict__ A_log,
                float* __restrict__ ys, int Lx, int ds) {
  const int bd = blockIdx.x;           // b*DI + d
  const int d  = bd & (DI_ - 1);
  const int bb = bd >> 8;
  const int s  = threadIdx.x;
  const bool act = (s < ds);
  const float A = act ? -__expf(A_log[d * ds + s]) : 0.f;
  float h = 0.f;
  for (int l = 0; l < Lx; ++l) {
    const size_t r = (size_t)(bb * Lx + l);
    const float dtv = dt[r * DI_ + d];
    const float xcv = xc[r * DI_ + d];
    const float Bm  = act ? xdb[r * XDB_LD + DTR + s]      : 0.f;
    const float Cm  = act ? xdb[r * XDB_LD + DTR + ds + s] : 0.f;
    h = __expf(dtv * A) * h + dtv * Bm * xcv;
    float yv = h * Cm;
#pragma unroll
    for (int off = 16; off >= 1; off >>= 1) yv += __shfl_xor(yv, off, 32);
    if (s == 0) ys[r * DI_ + d] = yv;
  }
}

// y = (ys + xc*D) * silu(z) ; z lives in xz cols [256,512)
__global__ void y_combine_k(const float* __restrict__ ys, const float* __restrict__ xc,
                            const float* __restrict__ Dp, const float* __restrict__ xz,
                            float* __restrict__ y, int total) {
  const int i = blockIdx.x * blockDim.x + threadIdx.x;
  if (i >= total) return;
  const int ch = i & (DI_ - 1);
  const int r  = i >> 8;
  const float z = xz[(size_t)r * (2 * DI_) + DI_ + ch];
  y[i] = (ys[i] + xc[i] * Dp[ch]) * silu_f(z);
}

// Conformer GLU: h1 (rows,512) -> a*sigmoid(g) (rows,256)
__global__ void glu_k(const float* __restrict__ h1, float* __restrict__ g, int total) {
  const int i = blockIdx.x * blockDim.x + threadIdx.x;
  if (i >= total) return;
  const int ch = i & (INNER_ - 1);
  const int r  = i >> 8;
  g[i] = h1[(size_t)r * (2 * INNER_) + ch] * sigm_f(h1[(size_t)r * (2 * INNER_) + INNER_ + ch]);
}

// Conformer depthwise conv K=31 (same-pad) + BN affine + silu
__global__ void dwconv_bn_silu_k(const float* __restrict__ x, const float* __restrict__ wgt,
                                 const float* __restrict__ wb, const float* __restrict__ bg,
                                 const float* __restrict__ bbias, const float* __restrict__ bm,
                                 const float* __restrict__ bv, float* __restrict__ y,
                                 int Lx, int total) {
  const int i = blockIdx.x * blockDim.x + threadIdx.x;
  if (i >= total) return;
  const int ch = i & (INNER_ - 1);
  const int t  = i >> 8;
  const int l  = t % Lx;
  const int bb = t / Lx;
  float acc = wb[ch];
#pragma unroll 4
  for (int k = 0; k < KDW; ++k) {
    const int ll = l + k - KDW / 2;
    if (ll >= 0 && ll < Lx) acc += x[((size_t)(bb * Lx + ll)) * INNER_ + ch] * wgt[ch * KDW + k];
  }
  acc = (acc - bm[ch]) * rsqrtf(bv[ch] + 1e-5f) * bg[ch] + bbias[ch];
  y[i] = silu_f(acc);
}

// out_cat[b, 0:L] = a[b] ; out_cat[b, L:2L] = c[b]
__global__ void concat_k(const float* __restrict__ a, const float* __restrict__ c,
                         float* __restrict__ dst, int total) {
  const int i = blockIdx.x * blockDim.x + threadIdx.x;
  if (i >= total) return;
  const int col = i % CH;
  const int r   = (i / CH) % (2 * LSEQ);
  const int bb  = i / (CH * 2 * LSEQ);
  dst[i] = (r < LSEQ) ? a[((size_t)bb * LSEQ + r) * CH + col]
                      : c[((size_t)bb * LSEQ + (r - LSEQ)) * CH + col];
}

// dst[b, l] = src[b, L + l]
__global__ void slice_k(const float* __restrict__ src, float* __restrict__ dst, int total) {
  const int i = blockIdx.x * blockDim.x + threadIdx.x;
  if (i >= total) return;
  const int col = i % CH;
  const int l   = (i / CH) % LSEQ;
  const int bb  = i / (CH * LSEQ);
  dst[i] = src[((size_t)bb * 2 * LSEQ + LSEQ + l) * CH + col];
}

__global__ void fill0_k(float* __restrict__ p, int total) {
  const int i = blockIdx.x * blockDim.x + threadIdx.x;
  if (i < total) p[i] = 0.f;
}

// ---------------------------------------------------------------------------
// Host side
// ---------------------------------------------------------------------------
namespace {

struct MP {
  const float *ln_w, *ln_b, *conv_w, *conv_b, *W_dt, *b_dt, *A_log, *D;
  const _Float16 *Win_p, *Wx_p, *Wout_p;   // fragment-packed weights
  int NxPad;                                // padded N for W_x
};

struct WS {
  float *skip1, *outfinal, *outf, *out1, *cat, *mout;
  float *ln, *xz, *xc, *xdb, *dt, *ys, *y;
};

inline dim3 gr(int n, int b = 256) { return dim3((n + b - 1) / b); }

inline void pack_w(const float* W, _Float16* Wp, int K, int N, int Npad, hipStream_t s) {
  const int total = Npad * K;   // (Npad/16)*(K/32)*32*16
  pack_w_k<<<gr(total), 256, 0, s>>>(W, Wp, K, N, total);
}

// Dispatch: NT chosen so Npad % (16*NT) == 0 for every shape we use.
inline void gemm(int K, const float* A, const _Float16* Wp, const float* bias, float* C,
                 int M, int Npad, int lda, int ldc, hipStream_t s) {
  dim3 b(32);
  if (K == 128) {                 // Npad in {512, 128}
    dim3 g(Npad / 64, M / 16);
    gemm_wmma_k<128, 4><<<g, b, 0, s>>>(A, Wp, bias, C, lda, ldc);
  } else if (Npad == 48) {
    dim3 g(1, M / 16);
    gemm_wmma_k<256, 3><<<g, b, 0, s>>>(A, Wp, bias, C, lda, ldc);
  } else if (Npad == 80) {
    dim3 g(1, M / 16);
    gemm_wmma_k<256, 5><<<g, b, 0, s>>>(A, Wp, bias, C, lda, ldc);
  } else {                        // Npad in {512, 128}
    dim3 g(Npad / 64, M / 16);
    gemm_wmma_k<256, 4><<<g, b, 0, s>>>(A, Wp, bias, C, lda, ldc);
  }
}

void run_mamba(const float* xin, int Lx, int ds, int dc, const MP& p,
               float* out, const WS& ws, hipStream_t s) {
  const int rows = BSZ * Lx;
  ln_fused_k<<<rows, 32, 0, s>>>(xin, nullptr, p.ln_w, p.ln_b, nullptr, ws.ln, 0);
  gemm(CH, ws.ln, p.Win_p, nullptr, ws.xz, rows, 2 * DI_, CH, 2 * DI_, s);
  const int tdi = rows * DI_;
  causal_conv_silu_k<<<gr(tdi), 256, 0, s>>>(ws.xz, p.conv_w, p.conv_b, ws.xc, Lx, dc, tdi);
  gemm(DI_, ws.xc, p.Wx_p, nullptr, ws.xdb, rows, p.NxPad, DI_, XDB_LD, s);
  dt_proj_k<<<gr(tdi), 256, 0, s>>>(ws.xdb, p.W_dt, p.b_dt, ws.dt, tdi);
  ssm_scan_k<<<BSZ * DI_, 32, 0, s>>>(ws.dt, ws.xdb, ws.xc, p.A_log, ws.ys, Lx, ds);
  y_combine_k<<<gr(tdi), 256, 0, s>>>(ws.ys, ws.xc, p.D, ws.xz, ws.y, tdi);
  gemm(DI_, ws.y, p.Wout_p, nullptr, out, rows, CH, DI_, CH, s);
}

} // namespace

extern "C" void kernel_launch(void* const* d_in, const int* in_sizes, int n_in,
                              void* d_out, int out_size, void* d_ws, size_t ws_size,
                              hipStream_t stream) {
  (void)in_sizes; (void)n_in; (void)out_size; (void)ws_size;
  const float* x_t0 = (const float*)d_in[0];

  const float* cv_ln_w = (const float*)d_in[23];
  const float* cv_ln_b = (const float*)d_in[24];
  const float* pw1_w   = (const float*)d_in[25];
  const float* pw1_b   = (const float*)d_in[26];
  const float* dw_w    = (const float*)d_in[27];
  const float* dw_b    = (const float*)d_in[28];
  const float* bn_g    = (const float*)d_in[29];
  const float* bn_b    = (const float*)d_in[30];
  const float* bn_m    = (const float*)d_in[31];
  const float* bn_v    = (const float*)d_in[32];
  const float* pw2_w   = (const float*)d_in[33];
  const float* pw2_b   = (const float*)d_in[34];
  const float* pn_w    = (const float*)d_in[35];
  const float* pn_b    = (const float*)d_in[36];

  // ---- workspace bump allocator ----
  float* p = (float*)d_ws;
  auto alloc = [&](size_t n) { float* r = p; p += n; return r; };
  WS ws;
  const int R1 = BSZ * LSEQ;           // 1024
  const int R2 = BSZ * 2 * LSEQ;       // 2048
  ws.skip1    = alloc((size_t)R1 * CH);
  ws.outfinal = alloc((size_t)R1 * CH);
  ws.outf     = alloc((size_t)R1 * CH);
  ws.out1     = alloc((size_t)R1 * CH);
  ws.cat      = alloc((size_t)R2 * CH);
  ws.mout     = alloc((size_t)R2 * CH);
  ws.ln       = alloc((size_t)R2 * CH);
  ws.xz       = alloc((size_t)R2 * 2 * DI_);
  ws.xc       = alloc((size_t)R2 * DI_);
  ws.xdb      = alloc((size_t)R2 * XDB_LD);
  ws.dt       = alloc((size_t)R2 * DI_);
  ws.ys       = alloc((size_t)R2 * DI_);
  ws.y        = alloc((size_t)R2 * DI_);

  _Float16* hp = (_Float16*)p;
  auto halloc = [&](size_t n) { _Float16* r = hp; hp += n; return r; };
  _Float16* Win1p = halloc((size_t)CH  * 512);
  _Float16* Wx1p  = halloc((size_t)DI_ * 48);
  _Float16* Wo1p  = halloc((size_t)DI_ * CH);
  _Float16* Win2p = halloc((size_t)CH  * 512);
  _Float16* Wx2p  = halloc((size_t)DI_ * 80);
  _Float16* Wo2p  = halloc((size_t)DI_ * CH);
  _Float16* Pw1p  = halloc((size_t)CH  * 512);
  _Float16* Pw2p  = halloc((size_t)DI_ * CH);

  // ---- pack weights (once per launch; deterministic, capture-safe) ----
  pack_w((const float*)d_in[3],  Win1p, CH,  512, 512, stream);  // m1.W_in
  pack_w((const float*)d_in[6],  Wx1p,  DI_, 40,  48,  stream);  // m1.W_x
  pack_w((const float*)d_in[11], Wo1p,  DI_, CH,  CH,  stream);  // m1.W_out
  pack_w((const float*)d_in[14], Win2p, CH,  512, 512, stream);  // m2.W_in
  pack_w((const float*)d_in[17], Wx2p,  DI_, 72,  80,  stream);  // m2.W_x
  pack_w((const float*)d_in[22], Wo2p,  DI_, CH,  CH,  stream);  // m2.W_out
  pack_w(pw1_w, Pw1p, CH,  512, 512, stream);
  pack_w(pw2_w, Pw2p, DI_, CH,  CH,  stream);

  auto mp = [&](int base, const _Float16* wi, const _Float16* wx, const _Float16* wo,
                int nxpad) {
    MP m;
    m.ln_w  = (const float*)d_in[base + 0];  m.ln_b  = (const float*)d_in[base + 1];
    m.conv_w= (const float*)d_in[base + 3];  m.conv_b= (const float*)d_in[base + 4];
    m.W_dt  = (const float*)d_in[base + 6];  m.b_dt  = (const float*)d_in[base + 7];
    m.A_log = (const float*)d_in[base + 8];  m.D     = (const float*)d_in[base + 9];
    m.Win_p = wi; m.Wx_p = wx; m.Wout_p = wo; m.NxPad = nxpad;
    return m;
  };
  const MP m1 = mp(1,  Win1p, Wx1p, Wo1p, 48);
  const MP m2 = mp(12, Win2p, Wx2p, Wo2p, 80);

  const size_t xbytes = (size_t)R1 * CH * sizeof(float);
  hipMemcpyAsync(ws.skip1,    x_t0, xbytes, hipMemcpyDeviceToDevice, stream);
  hipMemcpyAsync(ws.outfinal, x_t0, xbytes, hipMemcpyDeviceToDevice, stream);
  fill0_k<<<gr(R1 * CH), 256, 0, stream>>>(ws.outf, R1 * CH);

  for (int t = 0; t < 4; ++t) {
    // out = LN_pn(mamba1(LN(skip1))) + skip1
    run_mamba(ws.skip1, LSEQ, 16, 4, m1, ws.mout, ws, stream);
    ln_fused_k<<<R1, 32, 0, stream>>>(ws.mout, nullptr, pn_w, pn_b, ws.skip1, ws.out1, 2);

    // out_cat = [out_final ; out] along L, then mamba2 on 2L
    concat_k<<<gr(R2 * CH), 256, 0, stream>>>(ws.outfinal, ws.out1, ws.cat, R2 * CH);
    run_mamba(ws.cat, 2 * LSEQ, 32, 8, m2, ws.mout, ws, stream);

    // skip_full = LN_pn(out2) + out_cat (in place into cat); skip1 = second half
    ln_fused_k<<<R2, 32, 0, stream>>>(ws.mout, nullptr, pn_w, pn_b, ws.cat, ws.cat, 2);
    slice_k<<<gr(R1 * CH), 256, 0, stream>>>(ws.cat, ws.skip1, R1 * CH);

    // conformer_conv(skip1)
    ln_fused_k<<<R1, 32, 0, stream>>>(ws.skip1, nullptr, cv_ln_w, cv_ln_b, nullptr, ws.ln, 0);
    gemm(CH, ws.ln, Pw1p, pw1_b, ws.xz, R1, 2 * INNER_, CH, 2 * INNER_, stream);
    glu_k<<<gr(R1 * INNER_), 256, 0, stream>>>(ws.xz, ws.xc, R1 * INNER_);
    dwconv_bn_silu_k<<<gr(R1 * INNER_), 256, 0, stream>>>(ws.xc, dw_w, dw_b, bn_g, bn_b,
                                                          bn_m, bn_v, ws.y, LSEQ, R1 * INNER_);
    gemm(DI_, ws.y, Pw2p, pw2_b, ws.mout, R1, CH, DI_, CH, stream);

    // out_final = relu(LN_pn(conv_out + skip1)) ; out_f = LN_pn(out_f + out_final)
    ln_fused_k<<<R1, 32, 0, stream>>>(ws.mout, ws.skip1, pn_w, pn_b, nullptr, ws.outfinal, 1 | 4);
    ln_fused_k<<<R1, 32, 0, stream>>>(ws.outf, ws.outfinal, pn_w, pn_b, nullptr, ws.outf, 1);
  }

  hipMemcpyAsync(d_out, ws.outf, xbytes, hipMemcpyDeviceToDevice, stream);
}
